// Find_1898375545582
// MI455X (gfx1250) — compile-verified, matrix operand order checked by reading, not run on hardware
//
#include <hip/hip_runtime.h>

typedef __bf16 bf16;
typedef __attribute__((ext_vector_type(16))) __bf16 v16bf;
typedef __attribute__((ext_vector_type(8)))  __bf16 v8bf;
typedef __attribute__((ext_vector_type(8)))  float  v8f;

#define DIM   512
#define NSENT 1000
#define LSENT 50
#define NCAND 8
#define LCAND 32
#define LQ    16
#define TOPK  5
#define SROWS (NSENT*LSENT)
#define SPAD  (SROWS + 64)

// ---- output layout (floats), concatenated in reference return order ----
#define OUT_Q2P 0
#define OUT_C2P 512
#define OUT_V1  4608
#define OUT_S1  7168
#define OUT_V21 7173
#define OUT_V22 9733
#define OUT_S2  12293

// ---------------- WMMA helpers (wave32, 16x16x32 bf16) ----------------
__device__ __forceinline__ v8f wmma_bf16(v16bf a, v16bf b, v8f c) {
#if defined(__HIP_DEVICE_COMPILE__)
    return __builtin_amdgcn_wmma_f32_16x16x32_bf16(false, a, false, b, (short)0, c, false, false);
#else
    return c;
#endif
}

// A/B fragment load: lane holds row (lane&15); K chunks {h..h+7} and {h+16..h+23},
// h = lane<16 ? 0 : 8 (per CDNA5 ISA 16-bit A 16x32 layout; B mirrors A).
__device__ __forceinline__ v16bf frag_ld(const bf16* base, int row0, int ld, int kk, int lane) {
    int r = lane & 15;
    int h = (lane & 16) ? 8 : 0;
    const bf16* p = base + (size_t)(row0 + r) * ld + kk + h;
    union { v16bf v; v8bf q[2]; } u;
    u.q[0] = *(const v8bf*)p;
    u.q[1] = *(const v8bf*)(p + 16);
    return u.v;
}
__device__ __forceinline__ v16bf frag_ld_guard(const bf16* base, int row0, int ld, int kk, int lane, int rowlim) {
    int r = lane & 15;
    if (row0 + r >= rowlim) { v16bf z = {}; return z; }
    int h = (lane & 16) ? 8 : 0;
    const bf16* p = base + (size_t)(row0 + r) * ld + kk + h;
    union { v16bf v; v8bf q[2]; } u;
    u.q[0] = *(const v8bf*)p;
    u.q[1] = *(const v8bf*)(p + 16);
    return u.v;
}

// ---------------- prep kernels ----------------
// rows 0..15 = qvec rows, 16..271 = mc rows: dot with Sq_w/Sc_w + build bf16 A-mats.
__global__ void prep_qmc(const float* qvec, const float* mc,
                         const float* Sq_w, const float* Sc_w,
                         const float* Wq3, const float* Wc3,
                         const float* fq3, const float* fc3,
                         bf16* qA, bf16* qF, bf16* mcA, bf16* mcF,
                         float* sq_pre, float* sc_pre) {
    int row = blockIdx.x, tid = threadIdx.x;
    __shared__ float red[128];
    const float *src, *sw, *w3, *f3; bf16 *a, *f; float* pre;
    if (row < LQ) { src = qvec + (size_t)row*DIM; sw = Sq_w; w3 = Wq3; f3 = fq3;
                    a = qA + (size_t)row*DIM; f = qF + (size_t)row*DIM; pre = sq_pre + row; }
    else { int r = row - LQ; src = mc + (size_t)r*DIM; sw = Sc_w; w3 = Wc3; f3 = fc3;
           a = mcA + (size_t)r*DIM; f = mcF + (size_t)r*DIM; pre = sc_pre + r; }
    float part = 0.f;
    for (int d = tid; d < DIM; d += 128) {
        float v = src[d];
        part += v * sw[d];
        a[d] = (bf16)(v * w3[d]);
        f[d] = (bf16)(v * f3[d]);
    }
    red[tid] = part; __syncthreads();
    for (int s = 64; s > 0; s >>= 1) { if (tid < s) red[tid] += red[tid + s]; __syncthreads(); }
    if (tid == 0) *pre = red[0];
}

// one block per sentence row: bf16 convert + 4 dots (Wq2, Wc2, Ws2, Wm2)
__global__ void prep_sents(const float* sents, const float* Wq2, const float* Wc2,
                           const float* Ws2, const float* Wm2,
                           bf16* sents_bf, float* sdots) {
    int row = blockIdx.x, tid = threadIdx.x;
    __shared__ float red[4][128];
    if (row < SROWS) {
        float p0 = 0, p1 = 0, p2 = 0, p3 = 0;
        for (int d = tid; d < DIM; d += 128) {
            float v = sents[(size_t)row*DIM + d];
            sents_bf[(size_t)row*DIM + d] = (bf16)v;
            p0 += v*Wq2[d]; p1 += v*Wc2[d]; p2 += v*Ws2[d]; p3 += v*Wm2[d];
        }
        red[0][tid] = p0; red[1][tid] = p1; red[2][tid] = p2; red[3][tid] = p3;
        __syncthreads();
        for (int s = 64; s > 0; s >>= 1) {
            if (tid < s) for (int k = 0; k < 4; k++) red[k][tid] += red[k][tid + s];
            __syncthreads();
        }
        if (tid < 4) sdots[(size_t)row*4 + tid] = red[tid][0];
    } else {
        for (int d = tid; d < DIM; d += 128) sents_bf[(size_t)row*DIM + d] = (bf16)0.f;
        if (tid < 4) sdots[(size_t)row*4 + tid] = 0.f;
    }
}

__global__ void self_softmax(const float* sq_pre, const float* sc_pre,
                             const float* Sq_b, const float* Sc_b, const int* clens,
                             float* self_q, float* self_c) {
    int tid = threadIdx.x;
    if (tid == 0) {
        float b = Sq_b[0], mx = -3.4e38f, v[LQ];
        for (int i = 0; i < LQ; i++) { v[i] = sq_pre[i] + b; mx = fmaxf(mx, v[i]); }
        float sum = 0;
        for (int i = 0; i < LQ; i++) { v[i] = __expf(v[i] - mx); sum += v[i]; }
        for (int i = 0; i < LQ; i++) self_q[i] = v[i] / sum;
    }
    if (tid < NCAND) {
        int c = tid, cl = clens[c]; float b = Sc_b[0], mx = -3.4e38f;
        for (int l = 0; l < cl; l++) mx = fmaxf(mx, sc_pre[c*LCAND + l] + b);
        float sum = 0;
        for (int l = 0; l < cl; l++) { float e = __expf(sc_pre[c*LCAND + l] + b - mx); self_c[c*LCAND + l] = e; sum += e; }
        float inv = 1.f / sum;
        for (int l = 0; l < cl; l++) self_c[c*LCAND + l] *= inv;
        for (int l = cl; l < LCAND; l++) self_c[c*LCAND + l] = 0.f;
    }
}

// ---------------- context attention, query side (fused WMMA + softmax + Hq) ----------------
__global__ void ctx_q_kernel(const bf16* qA, const bf16* sents_bf, const float* sents_f,
                             const float* sdots, const float* self_q, const int* slens,
                             float* score_q, float* Hq) {
    int n = blockIdx.x, tid = threadIdx.x, wave = tid >> 5, lane = tid & 31;
    __shared__ float s[LQ][64];
    __shared__ float scq[64];
    int nt = wave;                       // 4 waves x 1 tile
    v8f acc = {};
    for (int kk = 0; kk < DIM; kk += 32) {
        v16bf a = frag_ld(qA, 0, DIM, kk, lane);
        v16bf b = frag_ld(sents_bf, n*LSENT + nt*16, DIM, kk, lane);
        acc = wmma_bf16(a, b, acc);
    }
    int col = lane & 15, mhi = (lane < 16) ? 0 : 8;
    for (int g = 0; g < 8; g++) s[g + mhi][nt*16 + col] = acc[g];
    __syncthreads();
    int slen = slens[n];
    if (tid < LQ) {                      // masked softmax over j per q (row terms cancel)
        int q = tid; float mx = -3.4e38f;
        for (int j = 0; j < slen; j++) { float v = s[q][j] + sdots[(size_t)(n*LSENT + j)*4 + 0]; s[q][j] = v; mx = fmaxf(mx, v); }
        float sum = 0;
        for (int j = 0; j < slen; j++) { float e = __expf(s[q][j] - mx); s[q][j] = e; sum += e; }
        float w = self_q[q] / sum;
        for (int j = 0; j < slen; j++) s[q][j] *= w;
        for (int j = slen; j < LSENT; j++) s[q][j] = 0.f;
    }
    __syncthreads();
    if (tid < LSENT) { float t = 0; for (int q = 0; q < LQ; q++) t += s[q][tid]; scq[tid] = t; score_q[n*LSENT + tid] = t; }
    __syncthreads();
    for (int d = tid; d < DIM; d += 128) {
        float a2 = 0;
        for (int j = 0; j < slen; j++) a2 += scq[j] * sents_f[((size_t)n*LSENT + j)*DIM + d];
        Hq[(size_t)n*DIM + d] = a2;
    }
}

// ---------------- context attention, candidate side (the 13 GFLOP GEMM, fused) ----------------
__global__ void ctx_c_kernel(const bf16* mcA, const bf16* sents_bf, const float* sents_f,
                             const float* sdots, const float* self_c, const int* slens,
                             const int* clens, float* Hc) {
    int n = blockIdx.x, c = blockIdx.y, tid = threadIdx.x, wave = tid >> 5, lane = tid & 31;
    __shared__ float s[LCAND][64];
    __shared__ float scc[64];
    for (int t = wave; t < 8; t += 4) {   // 2 Mtiles x 4 Ntiles
        int mt = t >> 2, nt = t & 3;
        v8f acc = {};
        for (int kk = 0; kk < DIM; kk += 32) {
            v16bf a = frag_ld(mcA, c*LCAND + mt*16, DIM, kk, lane);
            v16bf b = frag_ld(sents_bf, n*LSENT + nt*16, DIM, kk, lane);
            acc = wmma_bf16(a, b, acc);
        }
        int col = lane & 15, mhi = (lane < 16) ? 0 : 8;
        for (int g = 0; g < 8; g++) s[mt*16 + g + mhi][nt*16 + col] = acc[g];
    }
    __syncthreads();
    int slen = slens[n], clen = clens[c];
    if (tid < LCAND) {
        int l = tid;
        if (l < clen) {
            float mx = -3.4e38f;
            for (int j = 0; j < slen; j++) { float v = s[l][j] + sdots[(size_t)(n*LSENT + j)*4 + 1]; s[l][j] = v; mx = fmaxf(mx, v); }
            float sum = 0;
            for (int j = 0; j < slen; j++) { float e = __expf(s[l][j] - mx); s[l][j] = e; sum += e; }
            float w = self_c[c*LCAND + l] / sum;
            for (int j = 0; j < slen; j++) s[l][j] *= w;
            for (int j = slen; j < LSENT; j++) s[l][j] = 0.f;
        } else {
            for (int j = 0; j < LSENT; j++) s[l][j] = 0.f;
        }
    }
    __syncthreads();
    if (tid < LSENT) { float t = 0; for (int l = 0; l < LCAND; l++) t += s[l][tid]; scc[tid] = t; }
    __syncthreads();
    for (int d = tid; d < DIM; d += 128) {
        float a2 = 0;
        for (int j = 0; j < slen; j++) a2 += scc[j] * sents_f[((size_t)n*LSENT + j)*DIM + d];
        Hc[((size_t)c*NSENT + n)*DIM + d] = a2;
    }
}

// ---------------- generic H post-pass: bf16 convert (padded) + dot ----------------
__global__ void post_H(const float* H, const float* w, bf16* H_bf, float* dots, int nrows) {
    int row = blockIdx.x, tid = threadIdx.x;
    __shared__ float red[128];
    if (row < nrows) {
        float p = 0;
        for (int d = tid; d < DIM; d += 128) { float v = H[(size_t)row*DIM + d]; H_bf[(size_t)row*DIM + d] = (bf16)v; p += v*w[d]; }
        red[tid] = p; __syncthreads();
        for (int s = 64; s > 0; s >>= 1) { if (tid < s) red[tid] += red[tid + s]; __syncthreads(); }
        if (tid == 0) dots[row] = red[0];
    } else {
        for (int d = tid; d < DIM; d += 128) H_bf[(size_t)row*DIM + d] = (bf16)0.f;
        if (tid == 0) dots[row] = 0.f;
    }
}
__global__ void post_Hc(const float* Hc, const float* fc2, bf16* Hc_bf, float* Hcfc2) {
    int row = blockIdx.x, tid = threadIdx.x;
    int c = row >> 10, n = row & 1023;
    __shared__ float red[128];
    if (n < NSENT) {
        const float* src = Hc + ((size_t)c*NSENT + n)*DIM;
        bf16* dst = Hc_bf + (size_t)row*DIM;
        float p = 0;
        for (int d = tid; d < DIM; d += 128) { float v = src[d]; dst[d] = (bf16)v; p += v*fc2[d]; }
        red[tid] = p; __syncthreads();
        for (int s = 64; s > 0; s >>= 1) { if (tid < s) red[tid] += red[tid + s]; __syncthreads(); }
        if (tid == 0) Hcfc2[row] = red[0];
    } else {
        bf16* dst = Hc_bf + (size_t)row*DIM;
        for (int d = tid; d < DIM; d += 128) dst[d] = (bf16)0.f;
        if (tid == 0) Hcfc2[row] = 0.f;
    }
}

// ---------------- self/self2 attention hop (fused: LDS A build + WMMA + softmax + Hout) ----------------
__global__ void hop_kernel(const float* query, const float* att, const float* w3, int w2slot,
                           const bf16* sents_bf, const float* sents_f,
                           const float* sdots, const int* slens,
                           float* Hout, float* score_out) {
    int n = blockIdx.x, tid = threadIdx.x, wave = tid >> 5, lane = tid & 31;
    __shared__ bf16 A[LSENT * DIM];      // 51200 B
    __shared__ float s[LSENT][64];       // 12800 B
    __shared__ float scj[64];
    for (int idx = tid; idx < LSENT*DIM; idx += 256) {
        int i = idx >> 9, d = idx & (DIM - 1);
        A[idx] = (bf16)((query[(size_t)n*DIM + d] + sents_f[((size_t)n*LSENT + i)*DIM + d]) * w3[d]);
    }
    __syncthreads();
    for (int t = wave; t < 16; t += 8) { // 4x4 tiles, 8 waves
        int mt = t >> 2, nt = t & 3;
        v8f acc = {};
        for (int kk = 0; kk < DIM; kk += 32) {
            v16bf a = frag_ld_guard(A, mt*16, DIM, kk, lane, LSENT);
            v16bf b = frag_ld(sents_bf, n*LSENT + nt*16, DIM, kk, lane);
            acc = wmma_bf16(a, b, acc);
        }
        int col = lane & 15, mhi = (lane < 16) ? 0 : 8;
        for (int g = 0; g < 8; g++) {
            int row = mt*16 + g + mhi;
            if (row < LSENT) s[row][nt*16 + col] = acc[g];
        }
    }
    __syncthreads();
    int slen = slens[n];
    if (tid < LSENT) {
        int i = tid;
        if (i < slen) {                  // row term (qk@w1) cancels in softmax over j
            float mx = -3.4e38f;
            for (int j = 0; j < slen; j++) { float v = s[i][j] + sdots[(size_t)(n*LSENT + j)*4 + w2slot]; s[i][j] = v; mx = fmaxf(mx, v); }
            float sum = 0;
            for (int j = 0; j < slen; j++) { float e = __expf(s[i][j] - mx); s[i][j] = e; sum += e; }
            float w = att[n*LSENT + i] / sum;
            for (int j = 0; j < slen; j++) s[i][j] *= w;
            for (int j = slen; j < LSENT; j++) s[i][j] = 0.f;
        } else {
            for (int j = 0; j < LSENT; j++) s[i][j] = 0.f;   // att==0 on masked rows
        }
    }
    __syncthreads();
    if (tid < LSENT) { float t2 = 0; for (int i = 0; i < LSENT; i++) t2 += s[i][tid]; scj[tid] = t2; score_out[n*LSENT + tid] = t2; }
    __syncthreads();
    for (int d = tid; d < DIM; d += 256) {
        float a2 = 0;
        for (int j = 0; j < slen; j++) a2 += scj[j] * sents_f[((size_t)n*LSENT + j)*DIM + d];
        Hout[(size_t)n*DIM + d] = a2;
    }
}

// ---------------- find attention GEMMs ----------------
__global__ void fsq_gemm(const bf16* qF, const bf16* Hq_bf, float* fsq_raw) {
    int nt = blockIdx.x, lane = threadIdx.x;
    v8f acc = {};
    for (int kk = 0; kk < DIM; kk += 32) {
        v16bf a = frag_ld(qF, 0, DIM, kk, lane);
        v16bf b = frag_ld(Hq_bf, nt*16, DIM, kk, lane);
        acc = wmma_bf16(a, b, acc);
    }
    int col = lane & 15, mhi = (lane < 16) ? 0 : 8;
    for (int g = 0; g < 8; g++) fsq_raw[(size_t)(g + mhi)*1024 + nt*16 + col] = acc[g];
}
__global__ void fsc_gemm(const bf16* mcF, const bf16* Hc_bf, float* fsc_raw) {
    int nt = blockIdx.x, c = blockIdx.y, wave = threadIdx.x >> 5, lane = threadIdx.x & 31;
    v8f acc = {};
    for (int kk = 0; kk < DIM; kk += 32) {
        v16bf a = frag_ld(mcF + (size_t)c*LCAND*DIM, wave*16, DIM, kk, lane);
        v16bf b = frag_ld(Hc_bf + (size_t)c*1024*DIM, nt*16, DIM, kk, lane);
        acc = wmma_bf16(a, b, acc);
    }
    int col = lane & 15, mhi = (lane < 16) ? 0 : 8;
    for (int g = 0; g < 8; g++)
        fsc_raw[((size_t)c*LCAND + wave*16 + g + mhi)*1024 + nt*16 + col] = acc[g];
}

// softmax over 1000 cols of raw[row] + coldot; row-const terms cancel, no mask needed
__global__ void row_softmax_1000(const float* raw, int ldraw, const float* coldot,
                                 int gdiv, int gstride, float* out, int ldout) {
    int row = blockIdx.x, tid = threadIdx.x;
    __shared__ float red[256];
    const float* r = raw + (size_t)row * ldraw;
    const float* cd = coldot + (size_t)(row / gdiv) * gstride;
    float mx = -3.4e38f;
    for (int j = tid; j < NSENT; j += 256) mx = fmaxf(mx, r[j] + cd[j]);
    red[tid] = mx; __syncthreads();
    for (int s = 128; s > 0; s >>= 1) { if (tid < s) red[tid] = fmaxf(red[tid], red[tid + s]); __syncthreads(); }
    mx = red[0]; __syncthreads();
    float sum = 0;
    for (int j = tid; j < NSENT; j += 256) sum += __expf(r[j] + cd[j] - mx);
    red[tid] = sum; __syncthreads();
    for (int s = 128; s > 0; s >>= 1) { if (tid < s) red[tid] += red[tid + s]; __syncthreads(); }
    float inv = 1.f / red[0];
    for (int j = tid; j < NSENT; j += 256) out[(size_t)row*ldout + j] = __expf(r[j] + cd[j] - mx) * inv;
}

__global__ void qsc_kernel(const float* self_q, const float* fsq_p, float* q_sc) {
    int n = blockIdx.x * 256 + threadIdx.x;
    if (n < NSENT) {
        float t = 0;
        for (int q = 0; q < LQ; q++) t += self_q[q] * fsq_p[(size_t)q*NSENT + n];
        q_sc[n] = t;
    }
}
__global__ void csc_kernel(const float* self_c, const float* fsc_p, float* c_sc) {
    int n = blockIdx.x * 256 + threadIdx.x, c = blockIdx.y;
    if (n < NSENT) {
        float t = 0;
        for (int l = 0; l < LCAND; l++) t += self_c[c*LCAND + l] * fsc_p[((size_t)c*LCAND + l)*NSENT + n];
        c_sc[(size_t)c*NSENT + n] = t;
    }
}
__global__ void q2patt_kernel(const float* q_sc, const float* Hq, const float* self_q,
                              const float* qvec, float* q2p_att, float* self_qvec) {
    int d = threadIdx.x;
    float a = 0;
    for (int n = 0; n < NSENT; n++) a += q_sc[n] * Hq[(size_t)n*DIM + d];
    q2p_att[d] = a;
    float b = 0;
    for (int q = 0; q < LQ; q++) b += self_q[q] * qvec[(size_t)q*DIM + d];
    self_qvec[d] = b;
}
__global__ void c2patt_kernel(const float* c_sc, const float* Hc, const float* self_c,
                              const float* mc, float* c2p_att, float* self_cvec) {
    int c = blockIdx.x, d = threadIdx.x;
    float a = 0;
    for (int n = 0; n < NSENT; n++) a += c_sc[(size_t)c*NSENT + n] * Hc[((size_t)c*NSENT + n)*DIM + d];
    c2p_att[(size_t)c*DIM + d] = a;
    float b = 0;
    for (int l = 0; l < LCAND; l++) b += self_c[c*LCAND + l] * mc[((size_t)c*LCAND + l)*DIM + d];
    self_cvec[(size_t)c*DIM + d] = b;
}

// ---------------- wcom: [a, t, a*t] @ W^T + b ----------------
__global__ void wcom_kernel(const float* self_qvec, const float* q2p_att,
                            const float* self_cvec, const float* c2p_att,
                            const float* W, const float* bias,
                            float* q2p_ws, float* c2p_ws, float* out) {
    int row = blockIdx.x, tid = threadIdx.x;
    __shared__ float sa[DIM], st[DIM];
    const float* a = (row == 0) ? self_qvec : self_cvec + (size_t)(row - 1)*DIM;
    const float* t = (row == 0) ? q2p_att  : c2p_att  + (size_t)(row - 1)*DIM;
    for (int j = tid; j < DIM; j += 256) { sa[j] = a[j]; st[j] = t[j]; }
    __syncthreads();
    for (int d = tid; d < DIM; d += 256) {
        const float* wr = W + (size_t)d * (3*DIM);
        float acc = bias[d];
        for (int j = 0; j < DIM; j++)
            acc += sa[j]*wr[j] + st[j]*wr[DIM + j] + sa[j]*st[j]*wr[2*DIM + j];
        if (row == 0) { q2p_ws[d] = acc; out[OUT_Q2P + d] = acc; }
        else { c2p_ws[(size_t)(row - 1)*DIM + d] = acc; out[OUT_C2P + (row - 1)*DIM + d] = acc; }
    }
}

// ---------------- findm / findm2 ----------------
__global__ void s1_kernel(const float* Hm, const float* q2p, const float* Wfindm, float* s1_raw) {
    int n = blockIdx.x, tid = threadIdx.x;
    __shared__ float red[128];
    const float* wb = Wfindm + DIM; const float* wc = Wfindm + 2*DIM;
    float p = 0;
    for (int d = tid; d < DIM; d += 128) p += Hm[(size_t)n*DIM + d] * (wb[d] + q2p[d]*wc[d]);
    red[tid] = p; __syncthreads();
    for (int s = 64; s > 0; s >>= 1) { if (tid < s) red[tid] += red[tid + s]; __syncthreads(); }
    if (tid == 0) s1_raw[n] = red[0];
}

__global__ void findm_top(const float* s1_raw, const float* Hm,
                          float* score1_ws, float* v1_ws, float* out) {
    int tid = threadIdx.x;
    __shared__ float sm[NSENT];
    __shared__ float red[256];
    __shared__ float rv[256]; __shared__ int ri[256];
    __shared__ int ind[TOPK];
    float mx = -3.4e38f;
    for (int j = tid; j < NSENT; j += 256) mx = fmaxf(mx, s1_raw[j]);
    red[tid] = mx; __syncthreads();
    for (int s = 128; s > 0; s >>= 1) { if (tid < s) red[tid] = fmaxf(red[tid], red[tid + s]); __syncthreads(); }
    mx = red[0]; __syncthreads();
    float sum = 0;
    for (int j = tid; j < NSENT; j += 256) { float e = __expf(s1_raw[j] - mx); sm[j] = e; sum += e; }
    red[tid] = sum; __syncthreads();
    for (int s = 128; s > 0; s >>= 1) { if (tid < s) red[tid] += red[tid + s]; __syncthreads(); }
    float inv = 1.f / red[0];
    __syncthreads();
    for (int j = tid; j < NSENT; j += 256) sm[j] *= inv;
    __syncthreads();
    for (int k = 0; k < TOPK; k++) {
        float bv = -1.f; int bi = 0;
        for (int j = tid; j < NSENT; j += 256) { float v = sm[j]; if (v > bv) { bv = v; bi = j; } }
        rv[tid] = bv; ri[tid] = bi; __syncthreads();
        for (int s = 128; s > 0; s >>= 1) {
            if (tid < s) {
                if (rv[tid + s] > rv[tid] || (rv[tid + s] == rv[tid] && ri[tid + s] < ri[tid])) {
                    rv[tid] = rv[tid + s]; ri[tid] = ri[tid + s];
                }
            }
            __syncthreads();
        }
        if (tid == 0) { score1_ws[k] = rv[0]; out[OUT_S1 + k] = rv[0]; ind[k] = ri[0]; sm[ri[0]] = -1.f; }
        __syncthreads();
    }
    for (int idx = tid; idx < TOPK*DIM; idx += 256) {
        int k = idx >> 9, d = idx & (DIM - 1);
        float v = Hm[(size_t)ind[k]*DIM + d];
        v1_ws[idx] = v; out[OUT_V1 + idx] = v;
    }
}

__global__ void s2_kernel(const float* Hm2, const float* v1, const float* Wfindm2, float* s2_raw) {
    int n = blockIdx.x, tid = threadIdx.x;
    __shared__ float red[6][128];
    const float* wb = Wfindm2 + DIM; const float* wc = Wfindm2 + 2*DIM;
    float p[6] = {0, 0, 0, 0, 0, 0};
    for (int d = tid; d < DIM; d += 128) {
        float h = Hm2[(size_t)n*DIM + d];
        p[5] += h * wb[d];
        float hw = h * wc[d];
        for (int k = 0; k < TOPK; k++) p[k] += hw * v1[(size_t)k*DIM + d];
    }
    for (int k = 0; k < 6; k++) red[k][tid] = p[k];
    __syncthreads();
    for (int s = 64; s > 0; s >>= 1) {
        if (tid < s) for (int k = 0; k < 6; k++) red[k][tid] += red[k][tid + s];
        __syncthreads();
    }
    if (tid == 0) for (int k = 0; k < TOPK; k++) s2_raw[(size_t)k*NSENT + n] = red[k][0] + red[5][0];
}

__global__ void findm2_top(const float* s2_raw, const float* score1, const float* v1,
                           const float* Hm2, float* out) {
    int tid = threadIdx.x;
    __shared__ float comb[TOPK * NSENT];
    __shared__ float red[256];
    __shared__ float rv[256]; __shared__ int ri[256];
    __shared__ int i21[TOPK], i22[TOPK];
    for (int k = 0; k < TOPK; k++) {
        const float* r = s2_raw + (size_t)k*NSENT;
        float mx = -3.4e38f;
        for (int j = tid; j < NSENT; j += 256) mx = fmaxf(mx, r[j]);
        red[tid] = mx; __syncthreads();
        for (int s = 128; s > 0; s >>= 1) { if (tid < s) red[tid] = fmaxf(red[tid], red[tid + s]); __syncthreads(); }
        mx = red[0]; __syncthreads();
        float sum = 0;
        for (int j = tid; j < NSENT; j += 256) { float e = __expf(r[j] - mx); comb[k*NSENT + j] = e; sum += e; }
        red[tid] = sum; __syncthreads();
        for (int s = 128; s > 0; s >>= 1) { if (tid < s) red[tid] += red[tid + s]; __syncthreads(); }
        float f = score1[k] / red[0];
        __syncthreads();
        for (int j = tid; j < NSENT; j += 256) comb[k*NSENT + j] *= f;
        __syncthreads();
    }
    for (int k = 0; k < TOPK; k++) {
        float bv = -1.f; int bi = 0;
        for (int j = tid; j < TOPK*NSENT; j += 256) { float v = comb[j]; if (v > bv) { bv = v; bi = j; } }
        rv[tid] = bv; ri[tid] = bi; __syncthreads();
        for (int s = 128; s > 0; s >>= 1) {
            if (tid < s) {
                if (rv[tid + s] > rv[tid] || (rv[tid + s] == rv[tid] && ri[tid + s] < ri[tid])) {
                    rv[tid] = rv[tid + s]; ri[tid] = ri[tid + s];
                }
            }
            __syncthreads();
        }
        if (tid == 0) { out[OUT_S2 + k] = rv[0]; i21[k] = ri[0] / NSENT; i22[k] = ri[0] % NSENT; comb[ri[0]] = -1.f; }
        __syncthreads();
    }
    for (int idx = tid; idx < TOPK*DIM; idx += 256) {
        int k = idx >> 9, d = idx & (DIM - 1);
        out[OUT_V21 + idx] = v1[(size_t)i21[k]*DIM + d];
        out[OUT_V22 + idx] = Hm2[(size_t)i22[k]*DIM + d];
    }
}

// ---------------- host launch ----------------
extern "C" void kernel_launch(void* const* d_in, const int* in_sizes, int n_in,
                              void* d_out, int out_size, void* d_ws, size_t ws_size,
                              hipStream_t stream) {
    const float* qvec   = (const float*)d_in[0];
    const float* mc     = (const float*)d_in[1];
    const float* sents  = (const float*)d_in[2];
    const int*   clens  = (const int*)d_in[3];
    const int*   slens  = (const int*)d_in[4];
    const float* Wq2 = (const float*)d_in[6];
    const float* Wq3 = (const float*)d_in[7];
    const float* Wc2 = (const float*)d_in[9];
    const float* Wc3 = (const float*)d_in[10];
    const float* Ws2 = (const float*)d_in[12];
    const float* Ws3 = (const float*)d_in[13];
    const float* Wm2 = (const float*)d_in[15];
    const float* Wm3 = (const float*)d_in[16];
    const float* Sq_w = (const float*)d_in[17];
    const float* Sc_w = (const float*)d_in[18];
    const float* fq2 = (const float*)d_in[20];
    const float* fq3 = (const float*)d_in[21];
    const float* fc2 = (const float*)d_in[23];
    const float* fc3 = (const float*)d_in[24];
    const float* Sq_b = (const float*)d_in[25];
    const float* Sc_b = (const float*)d_in[26];
    const float* Wfindm  = (const float*)d_in[27];
    const float* Wfindm2 = (const float*)d_in[28];
    const float* Wcom_W  = (const float*)d_in[29];
    const float* Wcom_b  = (const float*)d_in[30];
    float* out = (float*)d_out;

    char* w = (char*)d_ws;
    size_t off = 0;
    auto alloc = [&](size_t bytes) -> void* { void* p = w + off; off = (off + bytes + 255) & ~(size_t)255; return p; };

    bf16*  sents_bf = (bf16*)alloc((size_t)SPAD * DIM * 2);
    float* sdots    = (float*)alloc((size_t)SPAD * 4 * 4);
    bf16*  qA  = (bf16*)alloc(LQ * DIM * 2);
    bf16*  qF  = (bf16*)alloc(LQ * DIM * 2);
    bf16*  mcA = (bf16*)alloc(NCAND * LCAND * DIM * 2);
    bf16*  mcF = (bf16*)alloc(NCAND * LCAND * DIM * 2);
    float* sq_pre = (float*)alloc(LQ * 4);
    float* sc_pre = (float*)alloc(NCAND * LCAND * 4);
    float* self_q = (float*)alloc(LQ * 4);
    float* self_c = (float*)alloc(NCAND * LCAND * 4);
    float* score_q  = (float*)alloc((size_t)SROWS * 4);
    float* Hq       = (float*)alloc((size_t)NSENT * DIM * 4);
    bf16*  Hq_bf    = (bf16*)alloc((size_t)1024 * DIM * 2);
    float* Hqfq2    = (float*)alloc(1024 * 4);
    float* Hc       = (float*)alloc((size_t)NCAND * NSENT * DIM * 4);
    bf16*  Hc_bf    = (bf16*)alloc((size_t)NCAND * 1024 * DIM * 2);
    float* Hcfc2    = (float*)alloc((size_t)NCAND * 1024 * 4);
    float* score_h1 = (float*)alloc((size_t)SROWS * 4);
    float* Hm       = (float*)alloc((size_t)NSENT * DIM * 4);
    float* score_h2 = (float*)alloc((size_t)SROWS * 4);
    float* Hm2      = (float*)alloc((size_t)NSENT * DIM * 4);
    float* fsq_raw  = (float*)alloc((size_t)LQ * 1024 * 4);
    float* fsq_p    = (float*)alloc((size_t)LQ * NSENT * 4);
    float* q_sc     = (float*)alloc(1024 * 4);
    float* q2p_att  = (float*)alloc(DIM * 4);
    float* self_qvec= (float*)alloc(DIM * 4);
    float* fsc_raw  = (float*)alloc((size_t)NCAND * LCAND * 1024 * 4);
    float* fsc_p    = (float*)alloc((size_t)NCAND * LCAND * NSENT * 4);
    float* c_sc     = (float*)alloc((size_t)NCAND * NSENT * 4);
    float* c2p_att  = (float*)alloc((size_t)NCAND * DIM * 4);
    float* self_cvec= (float*)alloc((size_t)NCAND * DIM * 4);
    float* q2p_ws   = (float*)alloc(DIM * 4);
    float* c2p_ws   = (float*)alloc((size_t)NCAND * DIM * 4);
    float* s1_raw   = (float*)alloc(1024 * 4);
    float* score1_ws= (float*)alloc(8 * 4);
    float* v1_ws    = (float*)alloc((size_t)TOPK * DIM * 4);
    float* s2_raw   = (float*)alloc((size_t)TOPK * NSENT * 4);
    (void)in_sizes; (void)n_in; (void)out_size; (void)ws_size;

    prep_qmc<<<LQ + NCAND*LCAND, 128, 0, stream>>>(qvec, mc, Sq_w, Sc_w, Wq3, Wc3, fq3, fc3,
                                                   qA, qF, mcA, mcF, sq_pre, sc_pre);
    prep_sents<<<SPAD, 128, 0, stream>>>(sents, Wq2, Wc2, Ws2, Wm2, sents_bf, sdots);
    self_softmax<<<1, 64, 0, stream>>>(sq_pre, sc_pre, Sq_b, Sc_b, clens, self_q, self_c);

    ctx_q_kernel<<<NSENT, 128, 0, stream>>>(qA, sents_bf, sents, sdots, self_q, slens, score_q, Hq);
    ctx_c_kernel<<<dim3(NSENT, NCAND), 128, 0, stream>>>(mcA, sents_bf, sents, sdots, self_c,
                                                         slens, clens, Hc);

    post_H<<<1024, 128, 0, stream>>>(Hq, fq2, Hq_bf, Hqfq2, NSENT);
    post_Hc<<<NCAND * 1024, 128, 0, stream>>>(Hc, fc2, Hc_bf, Hcfc2);

    hop_kernel<<<NSENT, 256, 0, stream>>>(Hq, score_q, Ws3, 2, sents_bf, sents, sdots, slens, Hm, score_h1);
    hop_kernel<<<NSENT, 256, 0, stream>>>(Hm, score_h1, Wm3, 3, sents_bf, sents, sdots, slens, Hm2, score_h2);

    fsq_gemm<<<64, 32, 0, stream>>>(qF, Hq_bf, fsq_raw);
    fsc_gemm<<<dim3(64, NCAND), 64, 0, stream>>>(mcF, Hc_bf, fsc_raw);

    row_softmax_1000<<<LQ, 256, 0, stream>>>(fsq_raw, 1024, Hqfq2, 1 << 28, 0, fsq_p, NSENT);
    row_softmax_1000<<<NCAND * LCAND, 256, 0, stream>>>(fsc_raw, 1024, Hcfc2, LCAND, 1024, fsc_p, NSENT);

    qsc_kernel<<<4, 256, 0, stream>>>(self_q, fsq_p, q_sc);
    csc_kernel<<<dim3(4, NCAND), 256, 0, stream>>>(self_c, fsc_p, c_sc);
    q2patt_kernel<<<1, 512, 0, stream>>>(q_sc, Hq, self_q, qvec, q2p_att, self_qvec);
    c2patt_kernel<<<NCAND, 512, 0, stream>>>(c_sc, Hc, self_c, mc, c2p_att, self_cvec);

    wcom_kernel<<<1 + NCAND, 256, 0, stream>>>(self_qvec, q2p_att, self_cvec, c2p_att,
                                               Wcom_W, Wcom_b, q2p_ws, c2p_ws, out);

    s1_kernel<<<NSENT, 128, 0, stream>>>(Hm, q2p_ws, Wfindm, s1_raw);
    findm_top<<<1, 256, 0, stream>>>(s1_raw, Hm, score1_ws, v1_ws, out);
    s2_kernel<<<NSENT, 128, 0, stream>>>(Hm2, v1_ws, Wfindm2, s2_raw);
    findm2_top<<<1, 256, 0, stream>>>(s2_raw, score1_ws, v1_ws, Hm2, out);
}
